// PatchAliBi_56959856280333
// MI455X (gfx1250) — matrix operand order repaired and checked by reading, not compiled, not run on hardware
//
#include <hip/hip_runtime.h>
#include <hip/hip_bf16.h>
#include <math.h>

typedef __attribute__((ext_vector_type(16))) _Float16 v16h;
typedef __attribute__((ext_vector_type(8)))  _Float16 v8h;
typedef __attribute__((ext_vector_type(8)))  float    v8f;

#define TOK   4096
#define SEQ   2048
#define BATCH 2
#define HID   512
#define NHEAD 8
#define HD    64
#define NLAYER 4

// ---------- fragment loaders ----------
// A fragment (16x32 f16, M=lane%16): halves i<8 -> K = 8*hi + i ; i>=8 -> K = 16 + 8*hi + (i-8)
// => two contiguous 8-half chunks at +0 and +16 halves from (row, k0 + 8*hi).
static __device__ __forceinline__ v16h load_frag_a(const _Float16* p) {
  v8h lo = *(const v8h*)p;
  v8h hi = *(const v8h*)(p + 16);
  return __builtin_shufflevector(lo, hi, 0,1,2,3,4,5,6,7,8,9,10,11,12,13,14,15);
}
// B fragment (32x16 f16, N=lane%16): halves i<8 -> K = 8*hi + i ; i>=8 -> K = 16 + 8*hi + (i-8).
static __device__ __forceinline__ v16h load_frag_b(const _Float16* p) {
  v8h lo = *(const v8h*)p;
  v8h hi = *(const v8h*)(p + 16);
  return __builtin_shufflevector(lo, hi, 0,1,2,3,4,5,6,7,8,9,10,11,12,13,14,15);
}

// ---------- elementwise f32 -> f16 ----------
__global__ void f32_to_f16_kernel(const float* __restrict__ in, _Float16* __restrict__ out, size_t n) {
  size_t i = (size_t)blockIdx.x * blockDim.x + threadIdx.x;
  size_t stride = (size_t)gridDim.x * blockDim.x;
  for (; i < n; i += stride) out[i] = (_Float16)in[i];
}

// ---------- LayerNorm (f32 in) -> f16 out, one 128-thread block per row of 512 ----------
__global__ void ln_to_f16_kernel(const float* __restrict__ x, const float* __restrict__ w,
                                 const float* __restrict__ b, _Float16* __restrict__ out) {
  __shared__ float rs[128];
  __shared__ float rs2[128];
  const int row = blockIdx.x;
  const int t = threadIdx.x;
  const float* xr = x + (size_t)row * HID;
  float4 v = ((const float4*)xr)[t];
  float s  = v.x + v.y + v.z + v.w;
  float s2 = v.x * v.x + v.y * v.y + v.z * v.z + v.w * v.w;
  rs[t] = s; rs2[t] = s2;
  __syncthreads();
  for (int off = 64; off > 0; off >>= 1) {
    if (t < off) { rs[t] += rs[t + off]; rs2[t] += rs2[t + off]; }
    __syncthreads();
  }
  const float mean = rs[0] * (1.0f / HID);
  const float var  = rs2[0] * (1.0f / HID) - mean * mean;
  const float rstd = rsqrtf(var + 1e-5f);
  _Float16* orow = out + (size_t)row * HID;
  const int c = t * 4;
  orow[c + 0] = (_Float16)((v.x - mean) * rstd * w[c + 0] + b[c + 0]);
  orow[c + 1] = (_Float16)((v.y - mean) * rstd * w[c + 1] + b[c + 1]);
  orow[c + 2] = (_Float16)((v.z - mean) * rstd * w[c + 2] + b[c + 2]);
  orow[c + 3] = (_Float16)((v.w - mean) * rstd * w[c + 3] + b[c + 3]);
}

// ---------- V transpose (LDS-tiled, coalesced both sides) ----------
// vT[(bh*64 + d)*SEQ + s] = qkv[(b*SEQ+s)*1536 + 1024 + h*64 + d]
__global__ void transpose_v_kernel(const _Float16* __restrict__ qkv, _Float16* __restrict__ vT) {
  __shared__ _Float16 tile[64][65];
  const int bh = blockIdx.y;            // 0..15
  const int b = bh >> 3, hh = bh & 7;
  const int s0 = blockIdx.x * 64;
  const int t = threadIdx.x;            // 0..255
#pragma unroll
  for (int i = 0; i < 16; ++i) {
    const int idx = i * 256 + t;        // 0..4095
    const int sl = idx >> 6, d = idx & 63;   // d contiguous -> coalesced read
    tile[d][sl] = qkv[(size_t)(b * SEQ + s0 + sl) * (3 * HID) + 2 * HID + hh * HD + d];
  }
  __syncthreads();
#pragma unroll
  for (int i = 0; i < 16; ++i) {
    const int idx = i * 256 + t;
    const int d = idx >> 6, sl = idx & 63;   // sl contiguous -> coalesced write
    vT[((size_t)bh * HD + d) * SEQ + s0 + sl] = tile[d][sl];
  }
}

// ---------- generic WMMA GEMM: C[M,N] = A[M,K] * W[N,K]^T (+bias, +gelu, +residual) ----------
// 4 waves per block; block tile 128(M) x 64(N); K step 32.
// W tile (64x32 f16, 4KB) staged into LDS once per block per k-step via
// CDNA5 async global->LDS copies (ASYNCcnt), double-buffered.
template<bool OUT_F16, bool RESID, bool GELU_>
__global__ void gemm_wmma_kernel(const _Float16* __restrict__ A, const _Float16* __restrict__ W,
                                 const float* __restrict__ bias,
                                 float* __restrict__ outF32, _Float16* __restrict__ outF16,
                                 int M, int N, int K) {
  __shared__ __align__(16) _Float16 btile[2][64 * 32];
  const int tid  = threadIdx.x;         // 0..127
  const int wav  = tid >> 5;            // 0..3
  const int lane = tid & 31;
  const int l16 = lane & 15, hi = lane >> 4;
  const int m0 = blockIdx.y * 128 + wav * 32;
  const int n0 = blockIdx.x * 64;

  // async-copy one 64(N) x 32(K) W tile into btile[buf]; 2 x b128 per thread
  auto issue_copy = [&](int buf, int k0) {
#pragma unroll
    for (int i = 0; i < 2; ++i) {
      const int u = tid + 128 * i;      // 0..255 chunk units of 16B
      const int row = u >> 2, chunk = u & 3;
      const _Float16* gp = W + (size_t)(n0 + row) * K + k0 + chunk * 8;
      unsigned lp = (unsigned)(size_t)&btile[buf][row * 32 + chunk * 8];
      asm volatile("global_load_async_to_lds_b128 %0, %1, off"
                   :: "v"(lp), "v"(gp) : "memory");
    }
  };

  v8f acc[2][4];
#pragma unroll
  for (int tm = 0; tm < 2; ++tm)
#pragma unroll
    for (int tn = 0; tn < 4; ++tn)
#pragma unroll
      for (int r = 0; r < 8; ++r) acc[tm][tn][r] = 0.0f;

  const int nk = K / 32;
  issue_copy(0, 0);
  for (int i = 0; i < nk; ++i) {
    if (i + 1 < nk) {
      issue_copy((i + 1) & 1, (i + 1) * 32);
      // previous buffer's 2 async loads complete in-order before the 2 just issued
      asm volatile("s_wait_asynccnt 0x2" ::: "memory");
    } else {
      asm volatile("s_wait_asynccnt 0x0" ::: "memory");
    }
    __syncthreads();
    const _Float16* bt = &btile[i & 1][0];
    v16h a[2], bfr[4];
#pragma unroll
    for (int tm = 0; tm < 2; ++tm)
      a[tm] = load_frag_a(A + (size_t)(m0 + tm * 16 + l16) * K + i * 32 + 8 * hi);
#pragma unroll
    for (int tn = 0; tn < 4; ++tn)
      bfr[tn] = load_frag_b(bt + (tn * 16 + l16) * 32 + 8 * hi);
#pragma unroll
    for (int tm = 0; tm < 2; ++tm)
#pragma unroll
      for (int tn = 0; tn < 4; ++tn)
        acc[tm][tn] = __builtin_amdgcn_wmma_f32_16x16x32_f16(
            false, a[tm], false, bfr[tn], (short)0, acc[tm][tn], false, false);
    __syncthreads();   // protect buffer (i&1) before it is overwritten at i+1's issue
  }

#pragma unroll
  for (int tm = 0; tm < 2; ++tm)
#pragma unroll
    for (int tn = 0; tn < 4; ++tn) {
      const int n = n0 + tn * 16 + l16;
      const float bv = bias ? bias[n] : 0.0f;
#pragma unroll
      for (int r = 0; r < 8; ++r) {
        const int m = m0 + tm * 16 + r + 8 * hi;
        float v = acc[tm][tn][r] + bv;
        if (GELU_) v = 0.5f * v * (1.0f + erff(v * 0.70710678118654752f));
        const size_t off = (size_t)m * N + n;
        if (OUT_F16) {
          outF16[off] = (_Float16)v;
        } else if (RESID) {
          outF32[off] = outF32[off] + v;
        } else {
          outF32[off] = v;
        }
      }
    }
}

// ---------- flash attention, ALiBi + causal; 1 wave per (b, head, 16-query tile) ----------
// No running max: post-LN scores are O(1) (clamped at +10 for safety), so exp never
// overflows f32 and P fits f16. All cross-lane reductions are deferred to a single
// row-sum reduction after the key loop -> no ds_bpermute chains in the inner loop.
__global__ void attn_kernel(const _Float16* __restrict__ qkv, const _Float16* __restrict__ vT,
                            _Float16* __restrict__ o16) {
  __shared__ __align__(16) _Float16 lds[16 * 32];
  const int lane = threadIdx.x;
  const int l16 = lane & 15, hi = lane >> 4;
  const int q0   = blockIdx.x * 16;
  const int head = blockIdx.y;
  const int b    = blockIdx.z;
  const float slope = exp2f(-(float)(head + 1));
  const float scale = 0.125f; // 1/sqrt(64)

  // Q fragments: 2 K-steps over hd=64 (one 16-row M tile)
  v16h qa[2];
#pragma unroll
  for (int ks = 0; ks < 2; ++ks) {
    const int s = q0 + l16;
    qa[ks] = load_frag_a(qkv + (size_t)(b * SEQ + s) * (3 * HID) + head * HD + ks * 32 + 8 * hi);
  }

  v8f o[4];
  float lpart[8];
#pragma unroll
  for (int t = 0; t < 4; ++t)
#pragma unroll
    for (int r = 0; r < 8; ++r) o[t][r] = 0.0f;
#pragma unroll
  for (int r = 0; r < 8; ++r) lpart[r] = 0.0f;

  for (int j0 = 0; j0 <= q0 + 15; j0 += 32) {
    // K^T B fragments: lane = key col, halves = contiguous hd
    v16h kb[2][2];
#pragma unroll
    for (int tn = 0; tn < 2; ++tn)
#pragma unroll
      for (int ks = 0; ks < 2; ++ks) {
        const int j = j0 + tn * 16 + l16;
        kb[tn][ks] = load_frag_b(qkv + (size_t)(b * SEQ + j) * (3 * HID) + HID + head * HD + ks * 32 + 8 * hi);
      }
    // scores: 16 q rows x 32 keys
    v8f sc[2];
#pragma unroll
    for (int tn = 0; tn < 2; ++tn) {
#pragma unroll
      for (int r = 0; r < 8; ++r) sc[tn][r] = 0.0f;
#pragma unroll
      for (int ks = 0; ks < 2; ++ks)
        sc[tn] = __builtin_amdgcn_wmma_f32_16x16x32_f16(
            false, qa[ks], false, kb[tn][ks], (short)0, sc[tn], false, false);
    }
    // scale + ALiBi + causal mask + exp ; accumulate per-lane partial row sums
#pragma unroll
    for (int tn = 0; tn < 2; ++tn) {
      const int j = j0 + tn * 16 + l16;
      const int diff = j - q0 - 8 * hi;       // j - q  when row offset r == 0
      const float base = slope * (float)diff; // -slope*(q-j) at r == 0
#pragma unroll
      for (int r = 0; r < 8; ++r) {
        float v = sc[tn][r] * scale + (base - slope * (float)r);
        v = fminf(v, 10.0f);
        if (diff > r) v = -3.0e38f;           // j > q -> masked
        const float p = __expf(v);
        sc[tn][r] = p;
        lpart[r] += p;
      }
    }
    // P: C-layout -> LDS -> A-layout fragment (16 rows x 32 keys)
#pragma unroll
    for (int tn = 0; tn < 2; ++tn)
#pragma unroll
      for (int r = 0; r < 8; ++r)
        lds[(r + 8 * hi) * 32 + tn * 16 + l16] = (_Float16)sc[tn][r];
    __syncthreads();
    v16h pa = load_frag_a(&lds[l16 * 32 + 8 * hi]);
    // O += P * V   (V B-fragment: lane = hd col, halves = contiguous keys from vT)
#pragma unroll
    for (int t = 0; t < 4; ++t) {
      const int d = t * 16 + l16;
      v16h vb = load_frag_b(vT + ((size_t)(b * NHEAD + head) * HD + d) * SEQ + j0 + 8 * hi);
      o[t] = __builtin_amdgcn_wmma_f32_16x16x32_f16(
          false, pa, false, vb, (short)0, o[t], false, false);
    }
    __syncthreads();
  }

  // one-time cross-lane row-sum reduction (16 lanes per row group)
#pragma unroll
  for (int r = 0; r < 8; ++r) {
    float s = lpart[r];
#pragma unroll
    for (int msk = 1; msk < 16; msk <<= 1)
      s += __shfl_xor(s, msk, 32);
    lpart[r] = s;
  }

  // normalize and write f16 attention output
#pragma unroll
  for (int t = 0; t < 4; ++t)
#pragma unroll
    for (int r = 0; r < 8; ++r) {
      const int s = q0 + r + 8 * hi;
      const int d = t * 16 + l16;
      const float val = o[t][r] / lpart[r];
      o16[(size_t)(b * SEQ + s) * HID + head * HD + d] = (_Float16)val;
    }
}

// ---------- host orchestration ----------
extern "C" void kernel_launch(void* const* d_in, const int* in_sizes, int n_in,
                              void* d_out, int out_size, void* d_ws, size_t ws_size,
                              hipStream_t stream) {
  (void)in_sizes; (void)n_in; (void)out_size; (void)ws_size;
  const float* x      = (const float*)d_in[0];
  const float* w_emb  = (const float*)d_in[1];
  const float* b_emb  = (const float*)d_in[2];
  const float* ln1_w  = (const float*)d_in[3];
  const float* ln1_b  = (const float*)d_in[4];
  const float* inp_w  = (const float*)d_in[5];
  const float* inp_b  = (const float*)d_in[6];
  const float* outp_w = (const float*)d_in[7];
  const float* outp_b = (const float*)d_in[8];
  const float* ln2_w  = (const float*)d_in[9];
  const float* ln2_b  = (const float*)d_in[10];
  const float* w1     = (const float*)d_in[11];
  const float* b1     = (const float*)d_in[12];
  const float* w2     = (const float*)d_in[13];
  const float* b2     = (const float*)d_in[14];
  const float* fln_w  = (const float*)d_in[15];
  const float* fln_b  = (const float*)d_in[16];
  const float* w_out  = (const float*)d_in[17];
  const float* b_out  = (const float*)d_in[18];
  float* out = (float*)d_out;

  // carve workspace (256B aligned)
  size_t off = 0;
  auto carve = [&](size_t bytes) -> void* {
    void* p = (char*)d_ws + off;
    off += (bytes + 255) & ~(size_t)255;
    return p;
  };
  _Float16* x16     = (_Float16*)carve((size_t)TOK * 192 * 2);
  _Float16* wemb16  = (_Float16*)carve((size_t)HID * 192 * 2);
  _Float16* inpw16  = (_Float16*)carve((size_t)NLAYER * 3 * HID * HID * 2);
  _Float16* outpw16 = (_Float16*)carve((size_t)NLAYER * HID * HID * 2);
  _Float16* w1_16   = (_Float16*)carve((size_t)NLAYER * 4 * HID * HID * 2);
  _Float16* w2_16   = (_Float16*)carve((size_t)NLAYER * HID * 4 * HID * 2);
  _Float16* wout16  = (_Float16*)carve((size_t)192 * HID * 2);
  _Float16* xn16    = (_Float16*)carve((size_t)TOK * HID * 2);
  _Float16* qkv16   = (_Float16*)carve((size_t)TOK * 3 * HID * 2);
  _Float16* vT16    = (_Float16*)carve((size_t)BATCH * NHEAD * HD * SEQ * 2);
  _Float16* o16     = (_Float16*)carve((size_t)TOK * HID * 2);
  _Float16* hid16   = (_Float16*)carve((size_t)TOK * 4 * HID * 2);
  float*    h       = (float*)carve((size_t)TOK * HID * 4);

  auto conv = [&](const float* src, _Float16* dst, size_t n) {
    int blocks = (int)((n + 255) / 256);
    f32_to_f16_kernel<<<blocks, 256, 0, stream>>>(src, dst, n);
  };
  conv(x,      x16,     (size_t)TOK * 192);
  conv(w_emb,  wemb16,  (size_t)HID * 192);
  conv(inp_w,  inpw16,  (size_t)NLAYER * 3 * HID * HID);
  conv(outp_w, outpw16, (size_t)NLAYER * HID * HID);
  conv(w1,     w1_16,   (size_t)NLAYER * 4 * HID * HID);
  conv(w2,     w2_16,   (size_t)NLAYER * HID * 4 * HID);
  conv(w_out,  wout16,  (size_t)192 * HID);

  const dim3 gblk(128);
  // embed: h = x @ w_emb^T + b_emb   [4096,192] x [512,192]^T -> f32
  gemm_wmma_kernel<false, false, false><<<dim3(HID / 64, TOK / 128), gblk, 0, stream>>>(
      x16, wemb16, b_emb, h, nullptr, TOK, HID, 192);

  for (int l = 0; l < NLAYER; ++l) {
    ln_to_f16_kernel<<<TOK, 128, 0, stream>>>(h, ln1_w + l * HID, ln1_b + l * HID, xn16);
    // qkv = xn @ inp_w^T + inp_b -> f16
    gemm_wmma_kernel<true, false, false><<<dim3(3 * HID / 64, TOK / 128), gblk, 0, stream>>>(
        xn16, inpw16 + (size_t)l * 3 * HID * HID, inp_b + l * 3 * HID,
        nullptr, qkv16, TOK, 3 * HID, HID);
    transpose_v_kernel<<<dim3(SEQ / 64, BATCH * NHEAD), 256, 0, stream>>>(qkv16, vT16);
    attn_kernel<<<dim3(SEQ / 16, NHEAD, BATCH), 32, 0, stream>>>(qkv16, vT16, o16);
    // h += o @ outp_w^T + outp_b
    gemm_wmma_kernel<false, true, false><<<dim3(HID / 64, TOK / 128), gblk, 0, stream>>>(
        o16, outpw16 + (size_t)l * HID * HID, outp_b + l * HID, h, nullptr, TOK, HID, HID);
    ln_to_f16_kernel<<<TOK, 128, 0, stream>>>(h, ln2_w + l * HID, ln2_b + l * HID, xn16);
    // hid = gelu(xn2 @ w1^T + b1) -> f16
    gemm_wmma_kernel<true, false, true><<<dim3(4 * HID / 64, TOK / 128), gblk, 0, stream>>>(
        xn16, w1_16 + (size_t)l * 4 * HID * HID, b1 + l * 4 * HID,
        nullptr, hid16, TOK, 4 * HID, HID);
    // h += hid @ w2^T + b2
    gemm_wmma_kernel<false, true, false><<<dim3(HID / 64, TOK / 128), gblk, 0, stream>>>(
        hid16, w2_16 + (size_t)l * HID * 4 * HID, b2 + l * HID, h, nullptr, TOK, HID, 4 * HID);
  }

  ln_to_f16_kernel<<<TOK, 128, 0, stream>>>(h, fln_w, fln_b, xn16);
  // out = hn @ w_out^T + b_out -> f32 d_out
  gemm_wmma_kernel<false, false, false><<<dim3(192 / 64, TOK / 128), gblk, 0, stream>>>(
      xn16, wout16, b_out, out, nullptr, TOK, 192, HID);
}
